// RecognitionRNN_20590073217308
// MI455X (gfx1250) — matrix operand order repaired and checked by reading, not compiled, not run on hardware
//
#include <hip/hip_runtime.h>
#include <hip/hip_bf16.h>

// ---------------------------------------------------------------------------
// RecognitionRNN forward on gfx1250 (MI455X), wave32 + WMMA bf16.
//   data  [B,T,OBS] f32, W_i2h [NHID, OBS+NHID] f32, b_i2h [NHID] f32,
//   W_h2o [LAT, NHID] f32, b_h2o [LAT] f32  ->  out [B,T,LAT] f32
// One wave owns 16 batch rows for all T; h kept in LDS (bf16), both weight
// matrices staged in LDS (bf16), f32 accum, hardware v_tanh_f32, pipelined
// x_t global loads. B-fragments stream from LDS through a ring-of-3 whose
// schedule is pinned by sched_group_barrier (2 DS-reads : 1 WMMA) so two
// ds_load_b128 pairs stay in flight across every v_wmma.
// ---------------------------------------------------------------------------

#define B_DIM   4096
#define T_DIM   256
#define OBS     64
#define NHID    128
#define LAT     32
#define CIN     (OBS + NHID)        // 192

#define W1_STRIDE   200             // halfwords; 400B rows -> conflict-free gather
#define W2_STRIDE   136             // halfwords; 272B rows
#define H_STRIDE    136             // halfwords; per-wave h tile [16][NHID]

#define WAVES_PER_BLOCK 4
#define ROWS_PER_WAVE   16
#define BLOCK_THREADS   (WAVES_PER_BLOCK * 32)

typedef __attribute__((ext_vector_type(16))) __bf16 v16bf;
typedef __attribute__((ext_vector_type(8)))  __bf16 v8bf;
typedef __attribute__((ext_vector_type(8)))  float  v8f;

// Scheduling groups: keep "2 DS reads, then 1 WMMA" pattern in the pipe.
__device__ __forceinline__ void sched_ds2_wmma1() {
#if __has_builtin(__builtin_amdgcn_sched_group_barrier)
    __builtin_amdgcn_sched_group_barrier(0x100, 2, 0);  // 2x DS read
    __builtin_amdgcn_sched_group_barrier(0x008, 1, 0);  // 1x MFMA/WMMA
#endif
}

// Hardware tanh (TRANS32 unit on CDNA5; co-executes with WMMA).
__device__ __forceinline__ float fast_tanh(float x) {
#if __has_builtin(__builtin_amdgcn_tanhf)
    return __builtin_amdgcn_tanhf(x);
#else
    float y;
    asm("v_tanh_f32 %0, %1" : "=v"(y) : "v"(x));
    return y;
#endif
}

// A fragment (16x32 bf16, MxK) from an LDS row-major tile.
// lane = hi*16 + m ; halfs j<8 -> K = k0 + hi*8 + j ; j>=8 -> K = k0+16+hi*8+(j-8)
__device__ __forceinline__ v16bf a_frag_lds(const __bf16* row, int k0, int hi) {
    const __bf16* p = row + k0 + hi * 8;
    v8bf lo = *(const v8bf*)(p);
    v8bf hv = *(const v8bf*)(p + 16);
    return __builtin_shufflevector(lo, hv, 0,1,2,3,4,5,6,7,8,9,10,11,12,13,14,15);
}

// A fragment from two raw f32 octets already resident in VGPRs.
__device__ __forceinline__ v16bf a_frag_from_f32(v8f lo, v8f hv) {
    v16bf r;
#pragma unroll
    for (int j = 0; j < 8; ++j) {
        r[j]     = (__bf16)lo[j];
        r[j + 8] = (__bf16)hv[j];
    }
    return r;
}

// B fragment (32x16 bf16, KxN) from LDS W stored row-major as W[n][k].
// lane = hi*16 + n ; halfs j<8 -> K = k0 + hi*8 + j ; j>=8 -> K = k0+16+hi*8+(j-8)
__device__ __forceinline__ v16bf b_frag_lds(const __bf16* w, int stride, int n0,
                                            int n, int hi, int k0) {
    const __bf16* p = w + (n0 + n) * stride + k0 + hi * 8;
    v8bf lo = *(const v8bf*)(p);
    v8bf hv = *(const v8bf*)(p + 16);
    return __builtin_shufflevector(lo, hv, 0,1,2,3,4,5,6,7,8,9,10,11,12,13,14,15);
}

__device__ __forceinline__ v8f wmma_bf16(v16bf a, v16bf b, v8f c) {
    return __builtin_amdgcn_wmma_f32_16x16x32_bf16(
        /*neg_a=*/false, a, /*neg_b=*/false, b,
        /*c_mod=*/(short)0, c, /*reuse_a=*/false, /*reuse_b=*/false);
}

__global__ void __launch_bounds__(BLOCK_THREADS)
rnn_wmma_kernel(const float* __restrict__ data,
                const float* __restrict__ Wi2h,
                const float* __restrict__ bi2h,
                const float* __restrict__ Wh2o,
                const float* __restrict__ bh2o,
                float* __restrict__ out) {
    extern __shared__ __align__(16) char smem_raw[];
    __bf16* sW1 = (__bf16*)smem_raw;                    // [NHID][W1_STRIDE]
    __bf16* sW2 = sW1 + NHID * W1_STRIDE;               // [LAT ][W2_STRIDE]
    __bf16* sH0 = sW2 + LAT * W2_STRIDE;                // [WAVES][16][H_STRIDE]

    const int tid = threadIdx.x;

    // ---- stage weights to LDS as bf16 (once) ----
    for (int i = tid; i < NHID * CIN; i += BLOCK_THREADS) {
        int r = i / CIN, c = i % CIN;
        sW1[r * W1_STRIDE + c] = (__bf16)Wi2h[i];
    }
    for (int i = tid; i < LAT * NHID; i += BLOCK_THREADS) {
        int r = i / NHID, c = i % NHID;
        sW2[r * W2_STRIDE + c] = (__bf16)Wh2o[i];
    }
    __syncthreads();

    const int wave = tid >> 5;
    const int lane = tid & 31;
    const int m    = lane & 15;   // A row within tile / B,C,D column within tile
    const int hi   = lane >> 4;

    __bf16* sH = sH0 + wave * 16 * H_STRIDE;            // this wave's h tile
    const int r0 = (blockIdx.x * WAVES_PER_BLOCK + wave) * ROWS_PER_WAVE;

    // ---- h0 = 0 (wave-private, DS ops in-order within a wave) ----
    for (int i = lane; i < 16 * NHID; i += 32) {
        int r = i / NHID, c = i % NHID;
        sH[r * H_STRIDE + c] = (__bf16)0.0f;
    }

    // ---- hoist per-lane biases (column n = nt*16 + m in C/D layout) ----
    float bias1[NHID / 16];
#pragma unroll
    for (int nt = 0; nt < NHID / 16; ++nt) bias1[nt] = bi2h[nt * 16 + m];
    float bias2[LAT / 16];
#pragma unroll
    for (int nt = 0; nt < LAT / 16; ++nt) bias2[nt] = bh2o[nt * 16 + m];

    // ---- software-pipelined x_t loads (raw f32 in VGPRs) ----
    const float* xbase = data + (size_t)(r0 + m) * (T_DIM * OBS);
    v8f xlo[OBS / 32], xhv[OBS / 32];
#pragma unroll
    for (int kc = 0; kc < OBS / 32; ++kc) {
        const float* p = xbase + kc * 32 + hi * 8;
        xlo[kc] = *(const v8f*)(p);
        xhv[kc] = *(const v8f*)(p + 16);
    }

    // B fragments by flat index
    auto b1_frag = [&](int idx) -> v16bf {   // idx = kc*8 + nt (i2h)
        int kc = idx >> 3, nt = idx & 7;
        return b_frag_lds(sW1, W1_STRIDE, nt * 16, m, hi, kc * 32);
    };
    auto b2_frag = [&](int idx) -> v16bf {   // idx = kc*2 + nt (h2o)
        int kc = idx >> 1, nt = idx & 1;
        return b_frag_lds(sW2, W2_STRIDE, nt * 16, m, hi, kc * 32);
    };

    // ---- time loop ----
    for (int t = 0; t < T_DIM; ++t) {
        // consume current x into bf16 A-fragments
        v16bf xa[OBS / 32];
#pragma unroll
        for (int kc = 0; kc < OBS / 32; ++kc)
            xa[kc] = a_frag_from_f32(xlo[kc], xhv[kc]);

        // issue next timestep's x loads early (clamped; overlaps WMMA body)
        {
            int tn = (t + 1 < T_DIM) ? (t + 1) : t;
            const float* row = xbase + tn * OBS;
#pragma unroll
            for (int kc = 0; kc < OBS / 32; ++kc) {
                const float* p = row + kc * 32 + hi * 8;
                xlo[kc] = *(const v8f*)(p);
                xhv[kc] = *(const v8f*)(p + 16);
            }
        }

        // i2h: acc[nt] = combined(16x192) @ W_i2h^T tile (+ bias)
        v8f acc[NHID / 16];
#pragma unroll
        for (int nt = 0; nt < NHID / 16; ++nt) {
            float bv = bias1[nt];
#pragma unroll
            for (int e = 0; e < 8; ++e) acc[nt][e] = bv;
        }

        // 48 WMMAs; ring-of-3 B pipeline (depth 2), schedule pinned so the
        // idx+2 fragment loads stay ABOVE wmma(idx).
        {
            v16bf a_cur  = xa[0];
            v16bf a_next = a_cur;
            v16bf bring[3];
            bring[0] = b1_frag(0);
            bring[1] = b1_frag(1);
#pragma unroll
            for (int idx = 0; idx < 48; ++idx) {
                const int kc = idx >> 3;
                const int nt = idx & 7;
                if (idx + 2 < 48) bring[(idx + 2) % 3] = b1_frag(idx + 2);
                if (nt == 0 && kc + 1 < CIN / 32) {       // prefetch next A chunk
                    int kn = kc + 1;
                    a_next = (kn < OBS / 32)
                                 ? xa[kn]
                                 : a_frag_lds(sH + m * H_STRIDE, kn * 32 - OBS, hi);
                }
                acc[nt] = wmma_bf16(a_cur, bring[idx % 3], acc[nt]);
                sched_ds2_wmma1();
                if (nt == 7) a_cur = a_next;
            }
        }

        // hardware tanh + write h_new back to LDS (C layout -> row-major)
#pragma unroll
        for (int nt = 0; nt < NHID / 16; ++nt) {
#pragma unroll
            for (int e = 0; e < 8; ++e) {
                float hv = fast_tanh(acc[nt][e]);
                int row = hi * 8 + e;                     // C/D row
                sH[row * H_STRIDE + nt * 16 + m] = (__bf16)hv;
            }
        }

        // h2o: 8 WMMAs; same ring-of-3 B pipeline + A chunk prefetch
        v8f oacc[LAT / 16];
#pragma unroll
        for (int nt = 0; nt < LAT / 16; ++nt) {
            float bv = bias2[nt];
#pragma unroll
            for (int e = 0; e < 8; ++e) oacc[nt][e] = bv;
        }
        {
            v16bf ha_cur  = a_frag_lds(sH + m * H_STRIDE, 0, hi);
            v16bf ha_next = ha_cur;
            v16bf b2ring[3];
            b2ring[0] = b2_frag(0);
            b2ring[1] = b2_frag(1);
#pragma unroll
            for (int idx = 0; idx < 8; ++idx) {
                const int kc = idx >> 1;
                const int nt = idx & 1;
                if (idx + 2 < 8) b2ring[(idx + 2) % 3] = b2_frag(idx + 2);
                if (nt == 0 && kc + 1 < NHID / 32)
                    ha_next = a_frag_lds(sH + m * H_STRIDE, (kc + 1) * 32, hi);
                oacc[nt] = wmma_bf16(ha_cur, b2ring[idx % 3], oacc[nt]);
                sched_ds2_wmma1();
                if (nt == 1) ha_cur = ha_next;
            }
        }

        // store out[b, t, :] : lane column n = nt*16 + m, row = hi*8 + e
#pragma unroll
        for (int nt = 0; nt < LAT / 16; ++nt) {
#pragma unroll
            for (int e = 0; e < 8; ++e) {
                int row = r0 + hi * 8 + e;
                out[(size_t)row * (T_DIM * LAT) + t * LAT + nt * 16 + m] = oacc[nt][e];
            }
        }
    }
}

extern "C" void kernel_launch(void* const* d_in, const int* in_sizes, int n_in,
                              void* d_out, int out_size, void* d_ws, size_t ws_size,
                              hipStream_t stream) {
    const float* data = (const float*)d_in[0];
    const float* Wi2h = (const float*)d_in[1];
    const float* bi2h = (const float*)d_in[2];
    const float* Wh2o = (const float*)d_in[3];
    const float* bh2o = (const float*)d_in[4];
    float* out = (float*)d_out;

    size_t shmem = (size_t)(NHID * W1_STRIDE + LAT * W2_STRIDE +
                            WAVES_PER_BLOCK * 16 * H_STRIDE) * sizeof(__bf16);
    dim3 grid(B_DIM / (WAVES_PER_BLOCK * ROWS_PER_WAVE));   // 64 blocks
    rnn_wmma_kernel<<<grid, BLOCK_THREADS, shmem, stream>>>(
        data, Wi2h, bi2h, Wh2o, bh2o, out);
}
